// DirectVoxGO_62173946577495
// MI455X (gfx1250) — matrix operand order repaired and checked by reading, not compiled, not run on hardware
//
#include <hip/hip_runtime.h>
#include <hip/hip_bf16.h>

typedef __attribute__((ext_vector_type(16))) _Float16 v16h;
typedef __attribute__((ext_vector_type(8)))  float    v8f;

#define GG    160
#define NSAMP 256
#define XSTR  72      // halfs per row of input stage (64 used, padded, 16B-aligned rows)
#define HSTR  136     // halfs per row of hidden stage (128 used, padded, 16B-aligned rows)
#define W0FRAGS (2 * 8 * 32 * 16)   // 8192 halfs
#define W1FRAGS (4 * 8 * 32 * 16)   // 16384 halfs
#define ACT_SHIFT_C (-13.815509557963774f)   // log(1e-6/(1-1e-6))
#define EPS_C 1e-10f

// ---------------------------------------------------------------------------
// Prep: swizzle w0 (pad K 39->64) and w1 into per-lane WMMA B-fragment layout.
//   B (32x16 f16): lanes 0-15 hold K=0..7,16..23, lanes 16-31 hold K=8..15,24..31 (elem e<8 -> K=8h+e, e>=8 -> K=16+8h+e-8)
//   column n = nt*16 + (lane & 15); frag index = (kt*8 + nt)*32 + lane
// ---------------------------------------------------------------------------
__global__ void dvgo_prep(const float* __restrict__ w0, const float* __restrict__ w1,
                          _Float16* __restrict__ wsw)
{
    int idx = blockIdx.x * blockDim.x + threadIdx.x;
    if (idx < W0FRAGS) {                              // w0: 2 ktiles x 8 ntiles
        int e    = idx & 15;
        int lane = (idx >> 4) & 31;
        int nt   = (idx >> 9) & 7;
        int kt   = idx >> 12;
        int n    = nt * 16 + (lane & 15);
        int k    = kt * 32 + ((lane >> 4) << 3) + (e & 7) + ((e >> 3) << 4);
        wsw[idx] = (k < 39) ? (_Float16)w0[k * 128 + n] : (_Float16)0.f;
    } else if (idx < W0FRAGS + W1FRAGS) {             // w1: 4 ktiles x 8 ntiles
        int i2   = idx - W0FRAGS;
        int e    = i2 & 15;
        int lane = (i2 >> 4) & 31;
        int nt   = (i2 >> 9) & 7;
        int kt   = i2 >> 12;
        int n    = nt * 16 + (lane & 15);
        int k    = kt * 32 + ((lane >> 4) << 3) + (e & 7) + ((e >> 3) << 4);
        wsw[i2 + W0FRAGS] = (_Float16)w1[k * 128 + n];
    }
}

// ---------------------------------------------------------------------------
// Main: one block = one ray (256 samples), 256 threads = 8 waves.
// ---------------------------------------------------------------------------
__global__ __launch_bounds__(256)
__attribute__((amdgpu_waves_per_eu(4)))
void dvgo_main(
    const float* __restrict__ ray_pts, const float* __restrict__ viewdirs,
    const float* __restrict__ dgrid,   const float* __restrict__ k0,
    const _Float16* __restrict__ wsw,
    const float* __restrict__ b0, const float* __restrict__ b1,
    const float* __restrict__ w2, const float* __restrict__ b2,
    float* __restrict__ out)
{
    __shared__ float s_vemb[27];
    __shared__ float s_scan[NSAMP];
    __shared__ float s_weights[NSAMP];
    __shared__ float s_Tlast;
    __shared__ float s_rgb[3];
    extern __shared__ _Float16 s_dyn[];
    _Float16* s_x    = s_dyn;                          // [256][XSTR]
    _Float16* s_hall = s_x + NSAMP * XSTR;             // 8 waves x [16][HSTR]
    _Float16* s_w0   = s_hall + 8 * 16 * HSTR;         // 8192 halfs
    _Float16* s_w1   = s_w0 + W0FRAGS;                 // 16384 halfs

    const int r    = blockIdx.x;
    const int t    = threadIdx.x;
    const int lane = t & 31;
    const int wv   = t >> 5;

    // ---- stage swizzled weights into LDS (48 KB, uint4 copies) ----
    {
        const uint4* src = (const uint4*)wsw;
        uint4*       dst = (uint4*)s_w0;
        #pragma unroll
        for (int i = 0; i < (W0FRAGS + W1FRAGS) / 8 / 256; ++i)   // 3072 uint4 / 256 thr = 12
            dst[t + i * 256] = src[t + i * 256];
    }

    if (t < 3) s_rgb[t] = 0.f;
    if (t == 0) {
        float vx = viewdirs[r * 3 + 0], vy = viewdirs[r * 3 + 1], vz = viewdirs[r * 3 + 2];
        float inv = 1.0f / (sqrtf(vx * vx + vy * vy + vz * vz) + EPS_C);
        float dv[3] = {vx * inv, vy * inv, vz * inv};
        s_vemb[0] = dv[0]; s_vemb[1] = dv[1]; s_vemb[2] = dv[2];
        #pragma unroll
        for (int i = 0; i < 3; ++i)
            #pragma unroll
            for (int j = 0; j < 4; ++j) {
                float ang = dv[i] * (float)(1 << j);
                s_vemb[3  + i * 4 + j] = __sinf(ang);   // v_sin_f32 path
                s_vemb[15 + i * 4 + j] = __cosf(ang);   // v_cos_f32 path
            }
    }

    // ---- sample point, trilinear setup ----
    float px = ray_pts[(r * NSAMP + t) * 3 + 0];
    float py = ray_pts[(r * NSAMP + t) * 3 + 1];
    float pz = ray_pts[(r * NSAMP + t) * 3 + 2];
    px = fminf(fmaxf(px, 0.f), 1.f) * (float)(GG - 1);
    py = fminf(fmaxf(py, 0.f), 1.f) * (float)(GG - 1);
    pz = fminf(fmaxf(pz, 0.f), 1.f) * (float)(GG - 1);
    int ix = (int)floorf(px); ix = ix < (GG - 2) ? ix : (GG - 2);
    int iy = (int)floorf(py); iy = iy < (GG - 2) ? iy : (GG - 2);
    int iz = (int)floorf(pz); iz = iz < (GG - 2) ? iz : (GG - 2);
    float fx = px - (float)ix, fy = py - (float)iy, fz = pz - (float)iz;
    float wx0 = 1.f - fx, wy0 = 1.f - fy, wz0 = 1.f - fz;
    int base = (ix * GG + iy) * GG + iz;

    // ---- density (C=1) trilinear -> alpha ----
    float d000 = dgrid[base],                d001 = dgrid[base + 1];
    float d010 = dgrid[base + GG],           d011 = dgrid[base + GG + 1];
    float d100 = dgrid[base + GG * GG],      d101 = dgrid[base + GG * GG + 1];
    float d110 = dgrid[base + GG * GG + GG], d111 = dgrid[base + GG * GG + GG + 1];
    float dens = wx0 * (wy0 * (wz0 * d000 + fz * d001) + fy * (wz0 * d010 + fz * d011))
               + fx  * (wy0 * (wz0 * d100 + fz * d101) + fy * (wz0 * d110 + fz * d111));
    float alpha = 1.f - rsqrtf(1.f + __expf(dens + ACT_SHIFT_C));   // power(-0.5), v_exp_f32 path

    // ---- block-wide inclusive cumprod of (1-alpha+eps) ----
    s_scan[t] = 1.f - alpha + EPS_C;
    __syncthreads();
    #pragma unroll
    for (int off = 1; off < NSAMP; off <<= 1) {
        float v = (t >= off) ? s_scan[t - off] : 1.f;
        __syncthreads();
        s_scan[t] *= v;
        __syncthreads();
    }
    float Texcl = (t == 0) ? 1.f : s_scan[t - 1];
    s_weights[t] = alpha * Texcl;
    if (t == NSAMP - 1) s_Tlast = s_scan[t];
    __syncthreads();

    // ---- k0 (C=12) trilinear ----
    float feat[12];
    #pragma unroll
    for (int c = 0; c < 12; ++c) feat[c] = 0.f;
    #pragma unroll
    for (int dx = 0; dx < 2; ++dx)
        #pragma unroll
        for (int dy = 0; dy < 2; ++dy)
            #pragma unroll
            for (int dz = 0; dz < 2; ++dz) {
                float w = (dx ? fx : wx0) * (dy ? fy : wy0) * (dz ? fz : wz0);
                const float4* cp = (const float4*)(k0 + (size_t)(base + dx * GG * GG + dy * GG + dz) * 12);
                float4 a = cp[0], b = cp[1], c4 = cp[2];
                feat[0] += w * a.x;  feat[1] += w * a.y;  feat[2]  += w * a.z;  feat[3]  += w * a.w;
                feat[4] += w * b.x;  feat[5] += w * b.y;  feat[6]  += w * b.z;  feat[7]  += w * b.w;
                feat[8] += w * c4.x; feat[9] += w * c4.y; feat[10] += w * c4.z; feat[11] += w * c4.w;
            }

    // ---- build padded f16 input row [feat(12) | vemb(27) | zeros -> 64] ----
    {
        _Float16* xr = s_x + t * XSTR;
        #pragma unroll
        for (int c = 0; c < 12; ++c) xr[c] = (_Float16)feat[c];
        #pragma unroll
        for (int j = 0; j < 27; ++j) xr[12 + j] = (_Float16)s_vemb[j];
        #pragma unroll
        for (int j = 39; j < 64; ++j) xr[j] = (_Float16)0.f;
    }

    // ---- per-wave MLP over this wave's two 16-sample M-tiles ----
    _Float16* hb = s_hall + wv * 16 * HSTR;
    const int rowA  = lane & 15;
    const int khalf = (lane >> 4) ? 8 : 0;
    const int rowC  = (lane >> 4) * 8;
    const int colC  = lane & 15;

    #pragma unroll 1
    for (int sub = 0; sub < 2; ++sub) {
        int mt   = wv * 2 + sub;
        int srow = mt * 16 + rowA;

        // layer 0: A frags from s_x (A 16x32 f16: elems 0-7 -> K khalf+0..7, 8-15 -> K khalf+16..23)
        v16h a0[2];
        #pragma unroll
        for (int kt = 0; kt < 2; ++kt) {
            const _Float16* p = s_x + srow * XSTR + kt * 32 + khalf;
            #pragma unroll
            for (int e = 0; e < 8; ++e) a0[kt][e] = p[e];
            #pragma unroll
            for (int e = 0; e < 8; ++e) a0[kt][8 + e] = p[16 + e];
        }
        #pragma unroll
        for (int nt = 0; nt < 8; ++nt) {
            v8f c = {};
            #pragma unroll
            for (int kt = 0; kt < 2; ++kt) {
                v16h b = *(const v16h*)(s_w0 + ((kt * 8 + nt) * 32 + lane) * 16);
                c = __builtin_amdgcn_wmma_f32_16x16x32_f16(false, a0[kt], false, b,
                                                           (short)0, c, false, false);
            }
            int col = nt * 16 + colC;
            float bias = b0[col];
            #pragma unroll
            for (int rr = 0; rr < 8; ++rr) {
                float h = fmaxf(c[rr] + bias, 0.f);
                hb[(rowC + rr) * HSTR + col] = (_Float16)h;
            }
        }

        // layer 1: A frags from hb (K = 128 -> 4 ktiles); load all before overwriting hb
        v16h a1[4];
        #pragma unroll
        for (int kt = 0; kt < 4; ++kt) {
            const _Float16* p = hb + rowA * HSTR + kt * 32 + khalf;
            #pragma unroll
            for (int e = 0; e < 8; ++e) a1[kt][e] = p[e];
            #pragma unroll
            for (int e = 0; e < 8; ++e) a1[kt][8 + e] = p[16 + e];
        }
        #pragma unroll
        for (int nt = 0; nt < 8; ++nt) {
            v8f c = {};
            #pragma unroll
            for (int kt = 0; kt < 4; ++kt) {
                v16h b = *(const v16h*)(s_w1 + ((kt * 8 + nt) * 32 + lane) * 16);
                c = __builtin_amdgcn_wmma_f32_16x16x32_f16(false, a1[kt], false, b,
                                                           (short)0, c, false, false);
            }
            int col = nt * 16 + colC;
            float bias = b1[col];
            #pragma unroll
            for (int rr = 0; rr < 8; ++rr) {
                float h = fmaxf(c[rr] + bias, 0.f);
                hb[(rowC + rr) * HSTR + col] = (_Float16)h;   // h1 overwrites h0 (already consumed)
            }
        }

        // layer 2 (128 -> 3) + sigmoid + render weight; lane halves split the K range
        {
            int jbase = (lane >> 4) * 64;
            float acc0 = 0.f, acc1 = 0.f, acc2 = 0.f;
            const _Float16* hr = hb + rowA * HSTR + jbase;
            #pragma unroll 8
            for (int j = 0; j < 64; ++j) {
                float hv = (float)hr[j];
                acc0 += hv * w2[(jbase + j) * 3 + 0];
                acc1 += hv * w2[(jbase + j) * 3 + 1];
                acc2 += hv * w2[(jbase + j) * 3 + 2];
            }
            acc0 += __shfl_xor(acc0, 16, 32);
            acc1 += __shfl_xor(acc1, 16, 32);
            acc2 += __shfl_xor(acc2, 16, 32);
            if (lane < 16) {
                float wgt = s_weights[mt * 16 + lane];
                atomicAdd(&s_rgb[0], wgt / (1.f + __expf(-(acc0 + b2[0]))));
                atomicAdd(&s_rgb[1], wgt / (1.f + __expf(-(acc1 + b2[1]))));
                atomicAdd(&s_rgb[2], wgt / (1.f + __expf(-(acc2 + b2[2]))));
            }
        }
    }

    __syncthreads();
    if (t < 3) out[r * 3 + t] = s_rgb[t] + s_Tlast;
}

// ---------------------------------------------------------------------------
extern "C" void kernel_launch(void* const* d_in, const int* in_sizes, int n_in,
                              void* d_out, int out_size, void* d_ws, size_t ws_size,
                              hipStream_t stream)
{
    const float* ray_pts  = (const float*)d_in[0];
    const float* viewdirs = (const float*)d_in[1];
    const float* dgrid    = (const float*)d_in[2];
    const float* k0       = (const float*)d_in[3];
    const float* w0       = (const float*)d_in[4];
    const float* b0       = (const float*)d_in[5];
    const float* w1       = (const float*)d_in[6];
    const float* b1       = (const float*)d_in[7];
    const float* w2       = (const float*)d_in[8];
    const float* b2       = (const float*)d_in[9];

    _Float16* wsw = (_Float16*)d_ws;   // [W0FRAGS | W1FRAGS] contiguous, 48 KB

    int NR = in_sizes[1] / 3;          // 8192 rays

    dvgo_prep<<<(W0FRAGS + W1FRAGS + 255) / 256, 256, 0, stream>>>(w0, w1, wsw);

    size_t dyn = (size_t)(NSAMP * XSTR + 8 * 16 * HSTR + W0FRAGS + W1FRAGS) * sizeof(_Float16); // 120832 B
    dvgo_main<<<NR, 256, dyn, stream>>>(ray_pts, viewdirs, dgrid, k0, wsw,
                                        b0, b1, w2, b2, (float*)d_out);
}
